// multiattention_67327907332625
// MI455X (gfx1250) — compile-verified, matrix operand order checked by reading, not run on hardware
//
#include <hip/hip_runtime.h>

// MI455X / gfx1250: f16 WMMA (fp32 accum) transformer block.
// Weights + activations pre-packed to f16 so GEMM hot loops are pure
// global_load_b128 + v_wmma_f32_16x16x32_f16 (no per-step conversions).

typedef __attribute__((ext_vector_type(16))) _Float16 v16h;
typedef __attribute__((ext_vector_type(8)))  float    v8f;

#define NB   8
#define NS   1024
#define ND   1024
#define NH   16
#define NDK  64
#define NL   2
#define NOUT 1024

#define WAVES_PER_BLOCK 8

// ---------------------------------------------------------------------------
// f16 fragment loader (CDNA5 16-bit A/B layout, wave32):
//   lanes 0-15 : row = lane,     halves 0-7 = K 0..7,  halves 8-15 = K 16..23
//   lanes 16-31: row = lane-16,  halves 0-7 = K 8..15, halves 8-15 = K 24..31
// B fragments load identically from row-major B^T (W[out,in], packed K, V^T).
// ---------------------------------------------------------------------------
__device__ __forceinline__ v16h frag_from_f16(const _Float16* tile, int ld, int lane) {
  const int r  = lane & 15;
  const int kb = (lane & 16) ? 8 : 0;
  const _Float16* p = tile + (size_t)r * ld + kb;
  v16h f;
#pragma unroll
  for (int i = 0; i < 8; ++i) f[i]     = p[i];
#pragma unroll
  for (int i = 0; i < 8; ++i) f[8 + i] = p[16 + i];
  return f;
}

__device__ __forceinline__ v8f wmma_f16(v16h a, v16h b, v8f c) {
  return __builtin_amdgcn_wmma_f32_16x16x32_f16(false, a, false, b, (short)0, c,
                                                false, false);
}

// ---------------------------------------------------------------------------
// fp32 -> f16 cast (weights once per call, layer-0 k/v activations).
// ---------------------------------------------------------------------------
__global__ __launch_bounds__(256) void cvt_kernel(const float* __restrict__ X,
                                                  _Float16* __restrict__ Y,
                                                  long n) {
  const long i = (long)blockIdx.x * 256 + threadIdx.x;
  if (i < n) Y[i] = (_Float16)X[i];
}

// ---------------------------------------------------------------------------
// LayerNorm: one block (256 thr) per row of D=1024; fp32 math,
// writes optional fp32 copy + f16 copy (consumed by WMMA GEMMs).
// ---------------------------------------------------------------------------
__global__ __launch_bounds__(256) void ln_kernel(const float* __restrict__ X,
                                                 const float* __restrict__ g,
                                                 const float* __restrict__ bta,
                                                 float* __restrict__ Y,
                                                 _Float16* __restrict__ Yh) {
  __shared__ float red[256];
  const int row = blockIdx.x;
  const int tid = threadIdx.x;
  const float* x = X + (size_t)row * ND;

  float s = 0.f;
  for (int i = tid; i < ND; i += 256) s += x[i];
  red[tid] = s; __syncthreads();
#pragma unroll
  for (int off = 128; off > 0; off >>= 1) {
    if (tid < off) red[tid] += red[tid + off];
    __syncthreads();
  }
  const float mean = red[0] * (1.f / ND);
  __syncthreads();

  float v = 0.f;
  for (int i = tid; i < ND; i += 256) { float d = x[i] - mean; v += d * d; }
  red[tid] = v; __syncthreads();
#pragma unroll
  for (int off = 128; off > 0; off >>= 1) {
    if (tid < off) red[tid] += red[tid + off];
    __syncthreads();
  }
  const float rstd = rsqrtf(red[0] * (1.f / ND) + 1e-8f);

  for (int i = tid; i < ND; i += 256) {
    const float yv = (x[i] - mean) * rstd * g[i] + bta[i];
    if (Y) Y[(size_t)row * ND + i] = yv;
    Yh[(size_t)row * ND + i] = (_Float16)yv;
  }
}

// ---------------------------------------------------------------------------
// GEMM: val = act( Xh[M,K] * Wh[N,K]^T + bias ) (+ fp32 resid)
// Block tile 64x128, 8 waves, 32x32 per wave -> 4 WMMA per K-step.
// Epilogue modes:
//   0: Y fp32 [M,N] (nullable) + Yh f16 [M,N] (nullable)
//   1: Yh f16 packed [B,H,S,64]   (Q/K head layout)
//   2: Yh f16 packed [B,H,64,S]   (V transposed head layout)
// ---------------------------------------------------------------------------
__global__ __launch_bounds__(256) void gemm_kernel(const _Float16* __restrict__ X,
                                                   const _Float16* __restrict__ W,
                                                   const float* __restrict__ bias,
                                                   const float* __restrict__ resid,
                                                   float* __restrict__ Y,
                                                   _Float16* __restrict__ Yh,
                                                   int M, int N, int K,
                                                   int relu, int mode) {
  const int wave = threadIdx.x >> 5;
  const int lane = threadIdx.x & 31;
  const int m0 = blockIdx.x * 64  + (wave & 1) * 32;
  const int n0 = blockIdx.y * 128 + (wave >> 1) * 32;

  const _Float16* a0 = X + (size_t)m0 * K;
  const _Float16* a1 = X + (size_t)(m0 + 16) * K;
  const _Float16* b0 = W + (size_t)n0 * K;
  const _Float16* b1 = W + (size_t)(n0 + 16) * K;

  v8f acc[2][2] = {};

  for (int kk = 0; kk < K; kk += 32) {
    if (kk + 32 < K) {  // gfx1250 global_prefetch_b8, next K-tile
      __builtin_prefetch(a0 + kk + 32, 0, 1);
      __builtin_prefetch(a1 + kk + 32, 0, 1);
      __builtin_prefetch(b0 + kk + 32, 0, 1);
      __builtin_prefetch(b1 + kk + 32, 0, 1);
    }
    const v16h fa0 = frag_from_f16(a0 + kk, K, lane);
    const v16h fa1 = frag_from_f16(a1 + kk, K, lane);
    const v16h fb0 = frag_from_f16(b0 + kk, K, lane);
    const v16h fb1 = frag_from_f16(b1 + kk, K, lane);
    acc[0][0] = wmma_f16(fa0, fb0, acc[0][0]);
    acc[0][1] = wmma_f16(fa0, fb1, acc[0][1]);
    acc[1][0] = wmma_f16(fa1, fb0, acc[1][0]);
    acc[1][1] = wmma_f16(fa1, fb1, acc[1][1]);
  }

  // C layout: vgpr r -> row (r + 8*(lane>=16)), col = lane&15
  const int mro  = (lane & 16) ? 8 : 0;
  const int ncol = lane & 15;
#pragma unroll
  for (int ma = 0; ma < 2; ++ma) {
#pragma unroll
    for (int nb = 0; nb < 2; ++nb) {
#pragma unroll
      for (int r = 0; r < 8; ++r) {
        const int m = m0 + 16 * ma + mro + r;
        const int n = n0 + 16 * nb + ncol;
        float yv = acc[ma][nb][r] + bias[n];
        if (relu) yv = fmaxf(yv, 0.f);
        if (resid) yv += resid[(size_t)m * N + n];
        if (mode == 0) {
          if (Y)  Y[(size_t)m * N + n] = yv;
          if (Yh) Yh[(size_t)m * N + n] = (_Float16)yv;
        } else {
          const int bb = m / NS, ss = m % NS;
          const int hh = n / NDK, dk = n % NDK;
          const size_t o = (mode == 1)
              ? ((((size_t)bb * NH + hh) * NS + ss) * NDK + dk)
              : ((((size_t)bb * NH + hh) * NDK + dk) * NS + ss);
          Yh[o] = (_Float16)yv;
        }
      }
    }
  }
}

// ---------------------------------------------------------------------------
// Flash-style attention. One wave = one (b, h, 16-query tile).
// Per 32-key chunk: 4 WMMA (QK^T) + online softmax via per-wave LDS +
// 4 WMMA (P*V). Wave-local ordering with s_wait_dscnt (CDNA5 split counters).
// ---------------------------------------------------------------------------
__global__ __launch_bounds__(256) void attn_kernel(const _Float16* __restrict__ Q16,
                                                   const _Float16* __restrict__ K16,
                                                   const _Float16* __restrict__ V16,
                                                   const int* __restrict__ mask,
                                                   _Float16* __restrict__ Outh) {
  __shared__ float    sS[WAVES_PER_BLOCK][16][32];
  __shared__ _Float16 sP[WAVES_PER_BLOCK][16][32];
  __shared__ float    sFac[WAVES_PER_BLOCK][16];
  __shared__ float    sSum[WAVES_PER_BLOCK][16];

  const int wave = threadIdx.x >> 5;
  const int lane = threadIdx.x & 31;
  const int gw = blockIdx.x * WAVES_PER_BLOCK + wave;
  const int qt = gw & 63;           // S/16 tiles
  const int h  = (gw >> 6) & (NH - 1);
  const int b  = gw >> 10;

  const float scale = 0.125f;       // 1/sqrt(64)

  const _Float16* qbase = Q16 + (((size_t)b * NH + h) * NS + (size_t)qt * 16) * NDK;
  const v16h qa0 = frag_from_f16(qbase,      NDK, lane);
  const v16h qa1 = frag_from_f16(qbase + 32, NDK, lane);

  v8f acc[4] = {};
  float rmax = -3.0e38f;   // lanes 0..15 own query row = lane
  float rsum = 0.f;

  const int mro  = (lane & 16) ? 8 : 0;
  const int ncol = lane & 15;

  for (int kc = 0; kc < NS; kc += 32) {
    // ---- scores: 16 queries x 32 keys ----
    const _Float16* kb = K16 + (((size_t)b * NH + h) * NS + kc) * NDK;
    const v16h k00 = frag_from_f16(kb,                 NDK, lane);
    const v16h k01 = frag_from_f16(kb + 32,            NDK, lane);
    const v16h k10 = frag_from_f16(kb + 16 * NDK,      NDK, lane);
    const v16h k11 = frag_from_f16(kb + 16 * NDK + 32, NDK, lane);
    v8f s0 = {};
    v8f s1 = {};
    s0 = wmma_f16(qa0, k00, s0);
    s0 = wmma_f16(qa1, k01, s0);
    s1 = wmma_f16(qa0, k10, s1);
    s1 = wmma_f16(qa1, k11, s1);

#pragma unroll
    for (int r = 0; r < 8; ++r) {
      sS[wave][mro + r][ncol]      = s0[r];
      sS[wave][mro + r][16 + ncol] = s1[r];
    }
    asm volatile("s_wait_dscnt 0" ::: "memory");

    // ---- online softmax: lane m (<16) owns query row m ----
    if (lane < 16) {
      const int m = lane;
      const int qrow = qt * 16 + m;
      const int* mrow = mask + ((size_t)b * NS + qrow) * NS + kc;
      float mx = rmax;
#pragma unroll
      for (int c = 0; c < 32; ++c) {
        float sc = sS[wave][m][c] * scale;
        sc = fminf(30.f, fmaxf(-30.f, sc));
        if (mrow[c] == 0) sc = -1e9f;
        sc += 1e-5f;
        sS[wave][m][c] = sc;
        mx = fmaxf(mx, sc);
      }
      const float f = __expf(rmax - mx);
      float sum = rsum * f;
#pragma unroll
      for (int c = 0; c < 32; ++c) {
        const float p = __expf(sS[wave][m][c] - mx);
        sum += p;
        sP[wave][m][c] = (_Float16)p;
      }
      sFac[wave][m] = f;
      rmax = mx;
      rsum = sum;
    }
    asm volatile("s_wait_dscnt 0" ::: "memory");

    // ---- rescale running accumulators ----
    float fr[8];
#pragma unroll
    for (int r = 0; r < 8; ++r) fr[r] = sFac[wave][mro + r];
#pragma unroll
    for (int nb = 0; nb < 4; ++nb)
#pragma unroll
      for (int r = 0; r < 8; ++r) acc[nb][r] *= fr[r];

    // ---- P * V : A-frag from LDS, B-frags from transposed V ----
    const v16h pa = frag_from_f16(&sP[wave][0][0], 32, lane);
    const _Float16* vb = V16 + (((size_t)b * NH + h) * NDK) * NS + kc;
#pragma unroll
    for (int nb = 0; nb < 4; ++nb) {
      const v16h vf = frag_from_f16(vb + (size_t)(nb * 16) * NS, NS, lane);
      acc[nb] = wmma_f16(pa, vf, acc[nb]);
    }
  }

  if (lane < 16) sSum[wave][lane] = rsum;
  asm volatile("s_wait_dscnt 0" ::: "memory");

  float inv[8];
#pragma unroll
  for (int r = 0; r < 8; ++r) inv[r] = 1.f / sSum[wave][mro + r];

#pragma unroll
  for (int nb = 0; nb < 4; ++nb) {
#pragma unroll
    for (int r = 0; r < 8; ++r) {
      const int q = qt * 16 + mro + r;
      const int d = h * NDK + nb * 16 + ncol;
      Outh[((size_t)b * NS + q) * ND + d] = (_Float16)(acc[nb][r] * inv[r]);
    }
  }
}

// ---------------------------------------------------------------------------
// Host orchestration
// ---------------------------------------------------------------------------
extern "C" void kernel_launch(void* const* d_in, const int* in_sizes, int n_in,
                              void* d_out, int out_size, void* d_ws, size_t ws_size,
                              hipStream_t stream) {
  const float* d_q   = (const float*)d_in[0];
  const float* d_k   = (const float*)d_in[1];
  const float* d_v   = (const float*)d_in[2];
  const int*   d_msk = (const int*)d_in[3];
  const float* Wqkv  = (const float*)d_in[4];
  const float* bqkv  = (const float*)d_in[5];
  const float* Wo    = (const float*)d_in[6];
  const float* bo    = (const float*)d_in[7];
  const float* ln1g  = (const float*)d_in[8];
  const float* ln1b  = (const float*)d_in[9];
  const float* ln2g  = (const float*)d_in[10];
  const float* ln2b  = (const float*)d_in[11];
  const float* Wc1   = (const float*)d_in[12];
  const float* bc1   = (const float*)d_in[13];
  const float* Wc2   = (const float*)d_in[14];
  const float* bc2   = (const float*)d_in[15];
  const float* lastg = (const float*)d_in[16];
  const float* lastb = (const float*)d_in[17];
  const float* Wfc   = (const float*)d_in[18];
  const float* bfc   = (const float*)d_in[19];
  float* out = (float*)d_out;

  const int M = NB * NS;                         // 8192
  const size_t slab = (size_t)NB * NS * ND;      // 8M elements

  // fp32 slabs
  float* bufX  = (float*)d_ws;       // attention-branch output (LN input)
  float* kmid  = bufX + slab;        // post-LN2 k (FFN residual)
  float* state = kmid + slab;        // end-of-layer k (fp32)
  // f16 activation slabs
  _Float16* lnH    = (_Float16*)(state + slab);
  _Float16* attnH  = lnH    + slab;
  _Float16* h1H    = attnH  + slab;
  _Float16* stateH = h1H    + slab;
  _Float16* k16in  = stateH + slab;
  _Float16* v16in  = k16in  + slab;
  _Float16* Q16    = v16in  + slab;  // [B,H,S,64]
  _Float16* K16    = Q16    + slab;  // [B,H,S,64]
  _Float16* V16    = K16    + slab;  // [B,H,64,S]
  // f16 weights
  _Float16* wqkvH = V16 + slab;                       // [L,3,D,D]
  _Float16* woH   = wqkvH + (size_t)NL * 3 * ND * ND; // [L,D,D]
  _Float16* wc1H  = woH   + (size_t)NL * ND * ND;
  _Float16* wc2H  = wc1H  + (size_t)NL * ND * ND;
  _Float16* wfcH  = wc2H  + (size_t)NL * ND * ND;     // [D_OUT,D]

  const dim3 blk(256);
  const dim3 gemm_grid(M / 64, ND / 128);
  const int  attn_grid = (NB * NH * (NS / 16)) / WAVES_PER_BLOCK;

  // --- pack weights + layer-0 k/v activations to f16 ---
  const long nWqkv = (long)NL * 3 * ND * ND;
  const long nWl   = (long)NL * ND * ND;
  const long nWfc  = (long)NOUT * ND;
  cvt_kernel<<<(int)(nWqkv / 256), blk, 0, stream>>>(Wqkv, wqkvH, nWqkv);
  cvt_kernel<<<(int)(nWl / 256),   blk, 0, stream>>>(Wo,   woH,   nWl);
  cvt_kernel<<<(int)(nWl / 256),   blk, 0, stream>>>(Wc1,  wc1H,  nWl);
  cvt_kernel<<<(int)(nWl / 256),   blk, 0, stream>>>(Wc2,  wc2H,  nWl);
  cvt_kernel<<<(int)(nWfc / 256),  blk, 0, stream>>>(Wfc,  wfcH,  nWfc);
  cvt_kernel<<<(int)(slab / 256),  blk, 0, stream>>>(d_k,  k16in, (long)slab);
  cvt_kernel<<<(int)(slab / 256),  blk, 0, stream>>>(d_v,  v16in, (long)slab);

  for (int i = 0; i < NL; ++i) {
    const float*    qs = i ? state  : d_q;
    const float*    ks = i ? state  : d_k;     // fp32 residual source
    const _Float16* kh = i ? stateH : k16in;   // f16 GEMM inputs
    const _Float16* vh = i ? stateH : v16in;
    const size_t wq = (size_t)i * 3 * ND * ND;
    const size_t bq = (size_t)i * 3 * ND;
    const size_t wl = (size_t)i * ND * ND;
    const size_t bl = (size_t)i * ND;

    // Q = LN(q); qh/kh/vh projections straight into packed head layouts
    ln_kernel<<<M, blk, 0, stream>>>(qs, ln1g + bl, ln1b + bl, nullptr, lnH);
    gemm_kernel<<<gemm_grid, blk, 0, stream>>>(lnH, wqkvH + wq,
        bqkv + bq, nullptr, nullptr, Q16, M, ND, ND, 0, 1);
    gemm_kernel<<<gemm_grid, blk, 0, stream>>>(kh, wqkvH + wq + (size_t)ND * ND,
        bqkv + bq + ND, nullptr, nullptr, K16, M, ND, ND, 0, 1);
    gemm_kernel<<<gemm_grid, blk, 0, stream>>>(vh, wqkvH + wq + 2 * (size_t)ND * ND,
        bqkv + bq + 2 * ND, nullptr, nullptr, V16, M, ND, ND, 0, 2);

    // attention
    attn_kernel<<<attn_grid, blk, 0, stream>>>(Q16, K16, V16, d_msk, attnH);

    // k = k + relu(Wo*attn + bo); k = LN(k)
    gemm_kernel<<<gemm_grid, blk, 0, stream>>>(attnH, woH + wl, bo + bl,
        ks, bufX, nullptr, M, ND, ND, 1, 0);
    ln_kernel<<<M, blk, 0, stream>>>(bufX, ln2g + bl, ln2b + bl, kmid, lnH);

    // k = k + Wc2*(Wc1*k + bc1) + bc2
    gemm_kernel<<<gemm_grid, blk, 0, stream>>>(lnH, wc1H + wl, bc1 + bl,
        nullptr, nullptr, h1H, M, ND, ND, 0, 0);
    gemm_kernel<<<gemm_grid, blk, 0, stream>>>(h1H, wc2H + wl, bc2 + bl,
        kmid, state, stateH, M, ND, ND, 0, 0);
  }

  // out = LN(k) * Wfc^T + bfc
  ln_kernel<<<M, blk, 0, stream>>>(state, lastg, lastb, nullptr, lnH);
  gemm_kernel<<<dim3(M / 64, NOUT / 128), blk, 0, stream>>>(lnH, wfcH, bfc,
      nullptr, out, nullptr, M, NOUT, ND, 0, 0);
}
